// MultiHeadedAttention_4612794876548
// MI455X (gfx1250) — matrix operand order repaired and checked by reading, not compiled, not run on hardware
//
#include <hip/hip_runtime.h>
#include <hip/hip_bf16.h>

// MI455X (gfx1250) wave32 WMMA implementation of multi-head attention.
// All heavy math goes through v_wmma_f32_16x16x32_f16 (f16 in, f32 accum).
// Softmax row reductions use DPP16 butterflies (VALU) instead of ds_bpermute,
// and exp runs in the log2 domain (v_exp_f32 is natively 2^x).

typedef __attribute__((ext_vector_type(16))) _Float16 v16h;
typedef __attribute__((ext_vector_type(8)))  _Float16 v8h;
typedef __attribute__((ext_vector_type(8)))  float    v8f;

#define B_   4
#define DIM_ 256
#define N_   2048
#define H_   4
#define DH_  64

// ---- DPP16 16-lane butterfly reductions (no LDS traffic) ----------------
// ctrl: 0xB1 = quad_perm(1,0,3,2) -> lane^1 ; 0x4E = quad_perm(2,3,0,1) -> ^2
//       0x141 = row_half_mirror -> ^7      ; 0x140 = row_mirror -> ^15
template <int CTRL>
__device__ __forceinline__ float dpp_mov(float x) {
  int i = __builtin_bit_cast(int, x);
  int r = __builtin_amdgcn_update_dpp(i, i, CTRL, 0xF, 0xF, false);
  return __builtin_bit_cast(float, r);
}
__device__ __forceinline__ float red_max16(float v) {
  v = fmaxf(v, dpp_mov<0xB1>(v));
  v = fmaxf(v, dpp_mov<0x4E>(v));
  v = fmaxf(v, dpp_mov<0x141>(v));
  v = fmaxf(v, dpp_mov<0x140>(v));
  return v;
}
__device__ __forceinline__ float red_sum16(float v) {
  v += dpp_mov<0xB1>(v);
  v += dpp_mov<0x4E>(v);
  v += dpp_mov<0x141>(v);
  v += dpp_mov<0x140>(v);
  return v;
}

// ---- fragment gathers (layouts per CDNA5 ISA 7.12.2) --------------------
// A (16x32 f16): lane holds row M = lane%16. half-slot j -> K =
//   (j<8 ? j : j+8) + (lane>=16 ? 8 : 0). Two contiguous 8-half (16B) loads.
__device__ __forceinline__ v16h load_a_frag(const _Float16* t, int stride,
                                            int row, int kb, int hi) {
  const _Float16* base = t + row * stride + kb + (hi ? 8 : 0);
  union { v16h v; v8h h[2]; } u;
  u.h[0] = *(const v8h*)(base);
  u.h[1] = *(const v8h*)(base + 16);
  return u.v;
}
// B (32x16 f16): lane holds column N = lane%16. half-slot j -> K =
//   (j<8 ? j : j+8) + (lane>=16 ? 8 : 0). Tile stored transposed [col][K]:
//   two contiguous 8-half (16B) loads.
__device__ __forceinline__ v16h load_b_frag(const _Float16* tT, int stride,
                                            int col, int kb, int hi) {
  const _Float16* base = tT + col * stride + kb + (hi ? 8 : 0);
  union { v16h v; v8h h[2]; } u;
  u.h[0] = *(const v8h*)(base);
  u.h[1] = *(const v8h*)(base + 16);
  return u.v;
}

// ---- 1x1-conv projection: y[b,o,n] = (sum_i W[o,i] x[b,i,n] + bias[o])*scale
// Workgroup (128 thr / 4 waves) computes a 64(o) x 64(n) tile; wave w owns
// o-rows [16w,16w+16). K staged in 32-wide blocks as f16 in LDS.
template <typename Tin, typename Tout>
__global__ void __launch_bounds__(128)
proj_kernel(const Tin* __restrict__ x, const float* __restrict__ W,
            const float* __restrict__ bias, Tout* __restrict__ y, float scale) {
  __shared__ __align__(16) _Float16 Ws[64 * 32];  // [o_rel][i_rel]
  __shared__ __align__(16) _Float16 Xs[64 * 32];  // [n_rel][i_rel] (transposed)
  const int tid  = threadIdx.x;
  const int lane = tid & 31, w = tid >> 5;
  const int hi = (lane >> 4) & 1, lc = lane & 15;
  const int b = blockIdx.z;
  const int oBase = blockIdx.y * 64;
  const int nBase = blockIdx.x * 64;

  v8f acc[4] = {};
  for (int kb = 0; kb < DIM_; kb += 32) {
    for (int l = tid; l < 64 * 32; l += 128) {
      int o = l >> 5, i = l & 31;
      Ws[o * 32 + i] = (_Float16)(W[(oBase + o) * DIM_ + kb + i]);
    }
    for (int l = tid; l < 32 * 64; l += 128) {
      int i = l >> 6, n = l & 63;  // consecutive tid -> consecutive n (coalesced)
      Xs[n * 32 + i] =
          (_Float16)((float)x[((size_t)b * DIM_ + kb + i) * N_ + nBase + n]);
    }
    __syncthreads();
    v16h a = load_a_frag(Ws, 32, w * 16 + lc, 0, hi);
#pragma unroll
    for (int nt = 0; nt < 4; ++nt) {
      v16h bf = load_b_frag(Xs, 32, nt * 16 + lc, 0, hi);
      acc[nt] = __builtin_amdgcn_wmma_f32_16x16x32_f16(
          false, a, false, bf, (short)0, acc[nt], false, false);
    }
    __syncthreads();
  }
#pragma unroll
  for (int r = 0; r < 8; ++r) {
    int o_rel = w * 16 + r + hi * 8;  // C frag: row = r + (hiLane?8:0)
    float bo = bias[oBase + o_rel];
#pragma unroll
    for (int nt = 0; nt < 4; ++nt) {
      int n = nBase + nt * 16 + lc;  // C frag: col = lane%16
      y[((size_t)b * DIM_ + oBase + o_rel) * N_ + n] =
          (Tout)((acc[nt][r] + bo) * scale);
    }
  }
}

// ---- flash attention per (b,h). 4 waves/WG; wave owns 16 query rows.
// Per 32-wide key block: 4 WMMAs for S (K=64 contraction), online softmax
// (f32 log2-domain, DPP row reductions), P restaged via LDS, 4 WMMAs for P*V.
__global__ void __launch_bounds__(128)
attn_kernel(const _Float16* __restrict__ Q, const _Float16* __restrict__ K,
            const _Float16* __restrict__ V, _Float16* __restrict__ X) {
  __shared__ __align__(16) _Float16 Qs[64 * 64];     // [n_rel][d]
  __shared__ __align__(16) _Float16 Ks[32 * 64];     // [m_rel][d] (B-frag for S)
  __shared__ __align__(16) _Float16 Vs[64 * 32];     // [d][m_rel] (B-frag for PV)
  __shared__ __align__(16) _Float16 Ps[4][16 * 32];  // per-wave P tile [n][m_rel]
  const int tid  = threadIdx.x;
  const int lane = tid & 31, w = tid >> 5;
  const int hi = (lane >> 4) & 1, lc = lane & 15;
  const int bh = blockIdx.y;
  const int b = bh >> 2, h = bh & 3;
  const int nBlk = blockIdx.x * 64;
  // channel c = 4*d + h  (reshape [B, D, H, N])
  const size_t base = ((size_t)b * DIM_ + h) * N_;

  for (int l = tid; l < 64 * 64; l += 128) {
    int d = l >> 6, n = l & 63;
    Qs[n * 64 + d] = Q[base + (size_t)4 * d * N_ + nBlk + n];
  }
  __syncthreads();
  // Q carries 1/sqrt(D) * log2(e), so scores are already in the log2 domain.
  v16h aq0 = load_a_frag(Qs, 64, w * 16 + lc, 0, hi);
  v16h aq1 = load_a_frag(Qs, 64, w * 16 + lc, 32, hi);

  v8f o0 = {}, o1 = {}, o2 = {}, o3 = {};
  float runm[8], runl[8];
#pragma unroll
  for (int r = 0; r < 8; ++r) { runm[r] = -1.0e30f; runl[r] = 0.0f; }

  for (int m0 = 0; m0 < N_; m0 += 32) {
    // prefetch next K/V block rows into cache (global_prefetch_b8)
    if (m0 + 32 < N_) {
      for (int l = tid; l < 64; l += 128) {
        size_t g = base + (size_t)4 * l * N_ + m0 + 32;
        __builtin_prefetch(&K[g], 0, 3);
        __builtin_prefetch(&V[g], 0, 3);
      }
    }
    for (int l = tid; l < 64 * 32; l += 128) {
      int d = l >> 5, m = l & 31;  // consecutive tid -> consecutive m
      size_t g = base + (size_t)4 * d * N_ + m0 + m;
      Ks[m * 64 + d] = K[g];
      Vs[d * 32 + m] = V[g];
    }
    __syncthreads();

    // S tiles: [16n x 16m] x2, contraction over d (two K=32 steps each)
    v8f s0 = {}, s1 = {};
    {
      v16h bk = load_b_frag(Ks, 64, lc, 0, hi);
      s0 = __builtin_amdgcn_wmma_f32_16x16x32_f16(false, aq0, false, bk, (short)0, s0, false, false);
      bk = load_b_frag(Ks, 64, lc, 32, hi);
      s0 = __builtin_amdgcn_wmma_f32_16x16x32_f16(false, aq1, false, bk, (short)0, s0, false, false);
      bk = load_b_frag(Ks, 64, 16 + lc, 0, hi);
      s1 = __builtin_amdgcn_wmma_f32_16x16x32_f16(false, aq0, false, bk, (short)0, s1, false, false);
      bk = load_b_frag(Ks, 64, 16 + lc, 32, hi);
      s1 = __builtin_amdgcn_wmma_f32_16x16x32_f16(false, aq1, false, bk, (short)0, s1, false, false);
    }

    // online softmax (log2 domain): row = r + hi*8 spans 16 lanes of a half-wave
#pragma unroll
    for (int r = 0; r < 8; ++r) {
      float v0 = s0[r], v1 = s1[r];
      float mr = red_max16(fmaxf(v0, v1));
      float nm = fmaxf(runm[r], mr);
      float corr = exp2f(runm[r] - nm);
      runm[r] = nm;
      float p0 = exp2f(v0 - nm), p1 = exp2f(v1 - nm);
      float rs = red_sum16(p0 + p1);
      runl[r] = runl[r] * corr + rs;
      o0[r] *= corr; o1[r] *= corr; o2[r] *= corr; o3[r] *= corr;
      int row = r + hi * 8;
      Ps[w][row * 32 + lc]      = (_Float16)p0;
      Ps[w][row * 32 + 16 + lc] = (_Float16)p1;
    }
    __syncthreads();

    // O += P[16x32] * V[32 x 64]  (4 d-tiles)
    v16h ap = load_a_frag(Ps[w], 32, lc, 0, hi);
    v16h bv = load_b_frag(Vs, 32, lc, 0, hi);
    o0 = __builtin_amdgcn_wmma_f32_16x16x32_f16(false, ap, false, bv, (short)0, o0, false, false);
    bv = load_b_frag(Vs, 32, 16 + lc, 0, hi);
    o1 = __builtin_amdgcn_wmma_f32_16x16x32_f16(false, ap, false, bv, (short)0, o1, false, false);
    bv = load_b_frag(Vs, 32, 32 + lc, 0, hi);
    o2 = __builtin_amdgcn_wmma_f32_16x16x32_f16(false, ap, false, bv, (short)0, o2, false, false);
    bv = load_b_frag(Vs, 32, 48 + lc, 0, hi);
    o3 = __builtin_amdgcn_wmma_f32_16x16x32_f16(false, ap, false, bv, (short)0, o3, false, false);
    __syncthreads();
  }

  // normalize and scatter X[b, c=4d+h, n] as f16
#pragma unroll
  for (int r = 0; r < 8; ++r) {
    float inv = 1.0f / runl[r];
    int n = nBlk + w * 16 + r + hi * 8;
    X[base + (size_t)4 * (lc)      * N_ + n] = (_Float16)(o0[r] * inv);
    X[base + (size_t)4 * (16 + lc) * N_ + n] = (_Float16)(o1[r] * inv);
    X[base + (size_t)4 * (32 + lc) * N_ + n] = (_Float16)(o2[r] * inv);
    X[base + (size_t)4 * (48 + lc) * N_ + n] = (_Float16)(o3[r] * inv);
  }
}

extern "C" void kernel_launch(void* const* d_in, const int* in_sizes, int n_in,
                              void* d_out, int out_size, void* d_ws, size_t ws_size,
                              hipStream_t stream) {
  const float* query = (const float*)d_in[0];
  const float* key_  = (const float*)d_in[1];
  const float* value = (const float*)d_in[2];
  const float* Wq = (const float*)d_in[3];
  const float* bq = (const float*)d_in[4];
  const float* Wk = (const float*)d_in[5];
  const float* bk = (const float*)d_in[6];
  const float* Wv = (const float*)d_in[7];
  const float* bv = (const float*)d_in[8];
  const float* Wo = (const float*)d_in[9];
  const float* bo = (const float*)d_in[10];
  float* out = (float*)d_out;

  // workspace: 4 f16 tensors of B*DIM*N (4 MB each, 16 MB total)
  const size_t elems = (size_t)B_ * DIM_ * N_;
  _Float16* Qh = (_Float16*)d_ws;
  _Float16* Kh = Qh + elems;
  _Float16* Vh = Kh + elems;
  _Float16* Xh = Vh + elems;

  dim3 gp(N_ / 64, DIM_ / 64, B_);
  // fold 1/sqrt(D) and log2(e) into the Q projection output:
  // softmax(s) == softmax_base2(s * log2e), and v_exp_f32 is natively 2^x.
  const float qscale = 0.125f * 1.44269504088896340736f;
  proj_kernel<float, _Float16><<<gp, 128, 0, stream>>>(query, Wq, bq, Qh, qscale);
  proj_kernel<float, _Float16><<<gp, 128, 0, stream>>>(key_,  Wk, bk, Kh, 1.0f);
  proj_kernel<float, _Float16><<<gp, 128, 0, stream>>>(value, Wv, bv, Vh, 1.0f);
  attn_kernel<<<dim3(N_ / 64, B_ * H_), 128, 0, stream>>>(Qh, Kh, Vh, Xh);
  proj_kernel<_Float16, float><<<gp, 128, 0, stream>>>(Xh, Wo, bo, out, 1.0f);
}
